// MultiHeadAttention_22488448762366
// MI455X (gfx1250) — matrix operand order repaired and checked by reading, not compiled, not run on hardware
//
#include <hip/hip_runtime.h>
#include <hip/hip_bf16.h>
#include <stdint.h>

// MI455X / gfx1250 multi-head attention, all matmuls via v_wmma_f32_16x16x32_f16.
// Pipeline: 3x projection GEMM -> flash attention (online softmax) -> output GEMM.
// WMMA fragment loads are lane-contiguous (2x ds_load_b128 per fragment).
// Pure-copy LDS staging uses CDNA5 async global->LDS DMA (ASYNCcnt), transposing
// or converting stages use packed v2h (b32) scatter stores.

typedef __attribute__((ext_vector_type(16))) _Float16 v16h;
typedef __attribute__((ext_vector_type(8)))  _Float16 v8h;
typedef __attribute__((ext_vector_type(2)))  _Float16 v2h;
typedef __attribute__((ext_vector_type(8)))  float    v8f;

#define B_  4
#define N_  2048
#define D_  1024
#define H_  16
#define DH_ 64

// ---- CDNA5 async global->LDS copy (16 bytes per lane), tracked by ASYNCcnt.
__device__ __forceinline__ void async_ld16(uint32_t lds_addr, const void* gptr) {
    asm volatile("global_load_async_to_lds_b128 %0, %1, off"
                 :: "v"(lds_addr), "v"((uint64_t)(uintptr_t)gptr)
                 : "memory");
}
__device__ __forceinline__ void wait_async0() {
    asm volatile("s_wait_asynccnt 0" ::: "memory");
}
// Low 32 bits of a generic pointer to __shared__ == LDS byte offset (aperture
// mapping truncates addr[31:0] into LDS space).
__device__ __forceinline__ uint32_t lds_off(const void* p) {
    return (uint32_t)(uintptr_t)p;
}

// Assemble a 16-half WMMA fragment from two contiguous 8-half (16B) LDS reads.
// A-matrix 16x32 f16 layout (ISA table): lane m holds K = {8g..8g+7, 16+8g..16+8g+7}.
// Here A and B both use contiguous K: lane holds K = {16g..16g+15} (one 32B run).
__device__ __forceinline__ v16h ld_frag(const _Float16* p0, const _Float16* p1) {
    v8h lo = *(const v8h*)p0;
    v8h hi = *(const v8h*)p1;
    return __builtin_shufflevector(lo, hi, 0, 1, 2, 3, 4, 5, 6, 7,
                                   8, 9, 10, 11, 12, 13, 14, 15);
}

// ---------------------------------------------------------------------------
// Tiled GEMM: C[M=8192, N=1024] = A[8192,1024] (f32 or f16) x W[1024,1024] (f32)
// Block tile 128x128, K-step 32, 256 threads = 8 waves, each wave does 32x64.
// Bs is stored [n][k] so B-fragment loads are K-contiguous per lane.
// HEAD_OUT: scatter f16 result into (B,H,N,DH); else write f32 row-major.
// f16 A-operand is staged with async global->LDS DMA (no conversion needed).
// ---------------------------------------------------------------------------
template <typename AT, bool HEAD_OUT>
__global__ void __launch_bounds__(256)
mha_gemm_kernel(const AT* __restrict__ A, const float* __restrict__ W,
                _Float16* __restrict__ outH, float* __restrict__ outF) {
    constexpr int Kd = D_;        // 1024
    constexpr int Nd = D_;        // 1024

    const int bm   = blockIdx.y * 128;
    const int bn   = blockIdx.x * 128;
    const int tid  = threadIdx.x;
    const int wave = tid >> 5;
    const int lane = tid & 31;
    const int nl   = lane & 15;   // N (B/C/D frags) or M (A frag)
    const int g    = lane >> 4;   // lane half-group
    const int wy   = wave & 3;    // 4 row strips of 32
    const int wx   = wave >> 2;   // 2 col strips of 64

    __shared__ __align__(16) _Float16 As[128][40];   // [m][k], 80B row stride
    __shared__ __align__(16) _Float16 Bs[128][40];   // [n][k] (transposed tile)

    v8f acc[2][4] = {};

    // A staging: 16 contiguous elements per thread from one row.
    const int ar  = tid >> 1;          // 0..127
    const int ac0 = (tid & 1) * 16;    // 0 / 16
    // B staging: 2 consecutive K-rows x 8 N-columns per thread, packed v2h scatter.
    const int bkp = tid >> 4;          // 0..15 -> k = {2*bkp, 2*bkp+1}
    const int bn0 = (tid & 15) * 8;    // 0..120

    const uint32_t a_lds = lds_off(&As[ar][ac0]);

    for (int k0 = 0; k0 < Kd; k0 += 32) {
        // Stage A tile.
        const AT* aptr = A + (size_t)(bm + ar) * Kd + k0 + ac0;
        if constexpr (sizeof(AT) == 2) {
            // Pure f16 copy: async DMA straight into LDS.
            async_ld16(a_lds, aptr);
            async_ld16(a_lds + 16, aptr + 8);
        } else {
            _Float16 tmp[16];
#pragma unroll
            for (int i = 0; i < 16; ++i) tmp[i] = (_Float16)aptr[i];
            *(v8h*)&As[ar][ac0]     = *(const v8h*)&tmp[0];
            *(v8h*)&As[ar][ac0 + 8] = *(const v8h*)&tmp[8];
        }
        // Stage W tile transposed into Bs[n][k]; packed b32 scatter stores.
        const float* bptr0 = W + (size_t)(k0 + 2 * bkp) * Nd + bn + bn0;
        const float* bptr1 = bptr0 + Nd;
#pragma unroll
        for (int i = 0; i < 8; ++i) {
            v2h p;
            p[0] = (_Float16)bptr0[i];
            p[1] = (_Float16)bptr1[i];
            *(v2h*)&Bs[bn0 + i][2 * bkp] = p;
        }
        if (k0 + 32 < Kd) {
            __builtin_prefetch(aptr + 32, 0, 1);              // global_prefetch_b8
            __builtin_prefetch(bptr0 + (size_t)32 * Nd, 0, 1);
        }
        if constexpr (sizeof(AT) == 2) wait_async0();
        __syncthreads();

        v16h af[2];
        v16h bf[4];
#pragma unroll
        for (int mt = 0; mt < 2; ++mt) {
            const _Float16* arow = &As[wy * 32 + mt * 16 + nl][0];
            af[mt] = ld_frag(arow + 16 * g, arow + 16 * g + 8);
        }
#pragma unroll
        for (int nt = 0; nt < 4; ++nt) {
            const _Float16* brow = &Bs[wx * 64 + nt * 16 + nl][0];
            bf[nt] = ld_frag(brow + 16 * g, brow + 16 * g + 8);
        }
#pragma unroll
        for (int mt = 0; mt < 2; ++mt)
#pragma unroll
            for (int nt = 0; nt < 4; ++nt)
                acc[mt][nt] = __builtin_amdgcn_wmma_f32_16x16x32_f16(
                    false, af[mt], false, bf[nt], (short)0, acc[mt][nt],
                    false, false);
        __syncthreads();
    }

    // Epilogue: D frag VGPR r -> M = 8*g + r, lane -> N = lane%16.
#pragma unroll
    for (int mt = 0; mt < 2; ++mt)
#pragma unroll
        for (int nt = 0; nt < 4; ++nt)
#pragma unroll
            for (int r = 0; r < 8; ++r) {
                const int row = bm + wy * 32 + mt * 16 + 8 * g + r;
                const int col = bn + wx * 64 + nt * 16 + nl;
                if constexpr (HEAD_OUT) {
                    const int bb = row >> 11, seq = row & (N_ - 1);
                    const int hh = col >> 6,  dh  = col & (DH_ - 1);
                    outH[(((size_t)(bb * H_ + hh)) * N_ + seq) * DH_ + dh] =
                        (_Float16)acc[mt][nt][r];
                } else {
                    outF[(size_t)row * Nd + col] = acc[mt][nt][r];
                }
            }
}

// ---------------------------------------------------------------------------
// Flash attention: one WG per (b, h, 64-query block); 4 waves of 32.
// Online softmax with per-row running max/sum; QK^T and P*V via WMMA.
// Q/K tiles staged with async global->LDS DMA; V staged transposed
// (Vts[dh][key]) with packed scatter so P*V B-fragments are K-contiguous.
// Q fragments are hoisted into registers for the whole key loop.
// ---------------------------------------------------------------------------
__global__ void __launch_bounds__(128)
mha_attn_kernel(const _Float16* __restrict__ Qp, const _Float16* __restrict__ Kp,
                const _Float16* __restrict__ Vp, const int* __restrict__ valid_lens,
                _Float16* __restrict__ Out) {
    const int qblk = blockIdx.x;
    const int h    = blockIdx.y;
    const int b    = blockIdx.z;
    const int tid  = threadIdx.x;
    const int wave = tid >> 5;
    const int lane = tid & 31;
    const int nl   = lane & 15;
    const int g    = lane >> 4;

    const int vlen = valid_lens[b];

    __shared__ __align__(16) _Float16 Qs[64][72];    // [q][dh]
    __shared__ __align__(16) _Float16 Ks[64][72];    // [key][dh]
    __shared__ __align__(16) _Float16 Vts[64][72];   // [dh][key] (transposed)
    __shared__ __align__(16) float    Ss[64][65];    // scores block
    __shared__ __align__(16) _Float16 Ps[64][72];    // probabilities
    __shared__ float mrow[64], lrow[64], crow[64];

    const size_t headbase = ((size_t)(b * H_ + h)) * N_ * DH_;

    // Row-major stagers: 32 contiguous f16 per thread = 4x async b128.
    const int lr  = tid >> 1;
    const int lc0 = (tid & 1) * 32;
    const uint32_t q_lds = lds_off(&Qs[lr][lc0]);
    const uint32_t k_lds = lds_off(&Ks[lr][lc0]);
    {
        const _Float16* qg = Qp + headbase + (size_t)(qblk * 64 + lr) * DH_ + lc0;
#pragma unroll
        for (int i = 0; i < 4; ++i) async_ld16(q_lds + 16 * i, qg + 8 * i);
    }
    if (tid < 64) { mrow[tid] = -1e30f; lrow[tid] = 0.0f; }

    // V transpose stager: 2 consecutive keys x 16 dh per thread, v2h scatter.
    const int vkp = tid >> 2;          // 0..31 -> keys {2*vkp, 2*vkp+1}
    const int vd0 = (tid & 3) * 16;    // 0..48

    wait_async0();
    __syncthreads();

    // Q fragments are loop-invariant: hoist into registers.
    v16h aqf[2];
#pragma unroll
    for (int kc = 0; kc < 2; ++kc) {
        const _Float16* qrow = &Qs[wave * 16 + nl][kc * 32];
        aqf[kc] = ld_frag(qrow + 16 * g, qrow + 16 * g + 8);
    }

    v8f oacc[4] = {};   // 16 rows x 64 dh per wave

    const int nkb = (vlen + 63) >> 6;   // skip fully-masked key blocks
    const float scale = 0.125f;         // 1/sqrt(DH)

    for (int kb = 0; kb < nkb; ++kb) {
        __syncthreads();   // previous iteration done with Ks/Vts/Ss/Ps
        const size_t kvbase = headbase + (size_t)(kb * 64) * DH_;
        {
            const _Float16* kg = Kp + kvbase + (size_t)lr * DH_ + lc0;
#pragma unroll
            for (int i = 0; i < 4; ++i) async_ld16(k_lds + 16 * i, kg + 8 * i);
        }
        {
            const _Float16* v0 = Vp + kvbase + (size_t)(2 * vkp) * DH_ + vd0;
            const _Float16* v1 = v0 + DH_;
#pragma unroll
            for (int i = 0; i < 16; ++i) {
                v2h p;
                p[0] = v0[i];
                p[1] = v1[i];
                *(v2h*)&Vts[vd0 + i][2 * vkp] = p;
            }
        }
        wait_async0();
        __syncthreads();

        // S = Q * K^T  (M=16 rows of this wave, N=64 keys, K=DH=64)
        v8f s[4] = {};
#pragma unroll
        for (int kc = 0; kc < 2; ++kc) {
#pragma unroll
            for (int nt = 0; nt < 4; ++nt) {
                const _Float16* krow = &Ks[nt * 16 + nl][kc * 32];
                v16h bk = ld_frag(krow + 16 * g, krow + 16 * g + 8);
                s[nt] = __builtin_amdgcn_wmma_f32_16x16x32_f16(
                    false, aqf[kc], false, bk, (short)0, s[nt], false, false);
            }
        }

        // Scale + mask, spill to LDS for the row-wise softmax pass.
#pragma unroll
        for (int nt = 0; nt < 4; ++nt)
#pragma unroll
            for (int r = 0; r < 8; ++r) {
                const int key = kb * 64 + nt * 16 + nl;
                float v = s[nt][r] * scale;
                if (key >= vlen) v = -1e20f;
                Ss[wave * 16 + 8 * g + r][nt * 16 + nl] = v;
            }
        __syncthreads();

        // Online softmax update, one thread per query row.
        if (tid < 64) {
            float bm = -1e30f;
#pragma unroll 8
            for (int j = 0; j < 64; ++j) bm = fmaxf(bm, Ss[tid][j]);
            const float om = mrow[tid];
            const float nm = fmaxf(om, bm);
            const float cr = __expf(om - nm);
            float ps = 0.0f;
#pragma unroll 8
            for (int j = 0; j < 64; ++j) {
                const float p = __expf(Ss[tid][j] - nm);
                ps += p;
                Ps[tid][j] = (_Float16)p;
            }
            lrow[tid] = lrow[tid] * cr + ps;
            mrow[tid] = nm;
            crow[tid] = cr;
        }
        __syncthreads();

        // Rescale running O, then O += P * V  (K dim = 64 keys).
        float cr8[8];
#pragma unroll
        for (int r = 0; r < 8; ++r) cr8[r] = crow[wave * 16 + 8 * g + r];
#pragma unroll
        for (int nt = 0; nt < 4; ++nt)
#pragma unroll
            for (int r = 0; r < 8; ++r) oacc[nt][r] *= cr8[r];

#pragma unroll
        for (int kc = 0; kc < 2; ++kc) {
            const _Float16* prow = &Ps[wave * 16 + nl][kc * 32];
            v16h ap = ld_frag(prow + 16 * g, prow + 16 * g + 8);
#pragma unroll
            for (int nt = 0; nt < 4; ++nt) {
                const _Float16* vrow = &Vts[nt * 16 + nl][kc * 32];
                v16h bv = ld_frag(vrow + 16 * g, vrow + 16 * g + 8);
                oacc[nt] = __builtin_amdgcn_wmma_f32_16x16x32_f16(
                    false, ap, false, bv, (short)0, oacc[nt], false, false);
            }
        }
    }

    // Normalize and store to (B, N, D) f16 (column = h*64 + dh).
    float linv[8];
#pragma unroll
    for (int r = 0; r < 8; ++r) linv[r] = 1.0f / lrow[wave * 16 + 8 * g + r];
#pragma unroll
    for (int nt = 0; nt < 4; ++nt)
#pragma unroll
        for (int r = 0; r < 8; ++r) {
            const int seq = qblk * 64 + wave * 16 + 8 * g + r;
            const int col = h * DH_ + nt * 16 + nl;
            Out[(size_t)(b * N_ + seq) * D_ + col] = (_Float16)(oacc[nt][r] * linv[r]);
        }
}

// ---------------------------------------------------------------------------
extern "C" void kernel_launch(void* const* d_in, const int* in_sizes, int n_in,
                              void* d_out, int out_size, void* d_ws, size_t ws_size,
                              hipStream_t stream) {
    (void)in_sizes; (void)n_in; (void)out_size; (void)ws_size;
    const float* Q  = (const float*)d_in[0];
    const float* K  = (const float*)d_in[1];
    const float* V  = (const float*)d_in[2];
    const int*   vl = (const int*)d_in[3];
    const float* Wq = (const float*)d_in[4];
    const float* Wk = (const float*)d_in[5];
    const float* Wv = (const float*)d_in[6];
    const float* Wo = (const float*)d_in[7];
    float* out = (float*)d_out;

    constexpr size_t SEG = (size_t)B_ * N_ * D_ * sizeof(_Float16);  // 16 MiB
    char* ws = (char*)d_ws;
    _Float16* qp = (_Float16*)(ws);
    _Float16* kp = (_Float16*)(ws + SEG);
    _Float16* vp = (_Float16*)(ws + 2 * SEG);
    _Float16* ao = (_Float16*)(ws + 3 * SEG);

    const dim3 ggrid(D_ / 128, (B_ * N_) / 128, 1);   // 8 x 64
    mha_gemm_kernel<float, true><<<ggrid, 256, 0, stream>>>(Q, Wq, qp, nullptr);
    mha_gemm_kernel<float, true><<<ggrid, 256, 0, stream>>>(K, Wk, kp, nullptr);
    mha_gemm_kernel<float, true><<<ggrid, 256, 0, stream>>>(V, Wv, vp, nullptr);

    mha_attn_kernel<<<dim3(N_ / 64, H_, B_), 128, 0, stream>>>(qp, kp, vp, vl, ao);

    mha_gemm_kernel<_Float16, false><<<ggrid, 256, 0, stream>>>(ao, Wo, nullptr, out);
}
